// OfficialVMambaEncoder_53154515256386
// MI455X (gfx1250) — matrix operand order repaired and checked by reading, not compiled, not run on hardware
//
// MI455X (gfx1250, CDNA5) VMamba encoder — compile-only optimization target.
//
// Design notes (reasoned from MI455X specs, no runtime available):
//  * All channel-mixing einsums (in_w, xproj, dt_w, out_w, fc1, fc2) are GEMMs
//    over L tokens -> lowered to v_wmma_f32_16x16x32_bf16 (f32 accumulate).
//  * GEMM tiles are DMA'd global->LDS with the Tensor Data Mover
//    (tensor_load_to_lds + s_wait_tensorcnt), double-buffered.
//  * Activations are token-major [L, C] f32 so LN / GEMM-A / scan all stream
//    contiguously; total working set < 192MB L2, so everything is L2-resident.
//  * Selective scan is a sequential per-channel recurrence over 4 direction
//    permutations (a tuned version would use chunked parallel scans).

#include <hip/hip_runtime.h>
#include <stdint.h>
#include <math.h>

#define DEV __device__ __forceinline__

typedef __attribute__((ext_vector_type(16))) __bf16 v16bf;
typedef __attribute__((ext_vector_type(8)))  float  v8f;
typedef __attribute__((ext_vector_type(4)))  unsigned int v4u;
typedef __attribute__((ext_vector_type(8)))  int    v8i;
typedef __attribute__((ext_vector_type(4)))  int    v4i;

// ---------------------------------------------------------------- helpers
DEV __bf16 f2bf(float f) {
  union { float f; unsigned u; } uf; uf.f = f;
  unsigned r = uf.u + 0x7fffu + ((uf.u >> 16) & 1u);   // RNE
  unsigned short h = (unsigned short)(r >> 16);
  union { unsigned short s; __bf16 b; } ub; ub.s = h; return ub.b;
}
DEV float geluf(float x)     { return 0.5f * x * (1.0f + erff(x * 0.70710678118654752f)); }
DEV float siluf(float x)     { return x / (1.0f + expf(-x)); }
DEV float softplusf(float x) { return (x > 20.f) ? x : log1pf(expf(x)); }
DEV v8f v8f_zero() { v8f z = {0.f,0.f,0.f,0.f,0.f,0.f,0.f,0.f}; return z; }

// Build a 2D TDM descriptor (D# group0 + group1 per CDNA5 ISA ch.8) and issue
// tensor_load_to_lds (6-arg toolchain variant: g0, g1, g2, g3, g4, cpol).
// Elements are bf16 (data_size=1 -> 2 bytes).  OOB rows / columns beyond
// (rem_rows, rem_k) are zero-filled by the TDM.
DEV void tdm_load_2d(unsigned lds_byte_addr, const void* gaddr,
                     unsigned tile_rows, unsigned tile_k,
                     unsigned rem_rows, unsigned rem_k, unsigned ld_elems)
{
  unsigned long long ga = (unsigned long long)(size_t)gaddr;
  v4u g0;
  g0[0] = 1u;                                   // count=1, user descriptor
  g0[1] = lds_byte_addr;                        // lds_addr
  g0[2] = (unsigned)(ga & 0xffffffffull);       // global_addr[31:0]
  g0[3] = (unsigned)((ga >> 32) & 0x01ffffffull) | 0x80000000u; // addr[56:32] | type=2
  union { unsigned long long q[4]; v8i v; } g1;
  unsigned long long d0  = rem_k;               // tensor_dim0 (bits 79:48)
  unsigned long long d1  = rem_rows;            // tensor_dim1 (bits 111:80)
  unsigned long long st0 = ld_elems;            // tensor_dim0_stride (bits 207:160)
  g1.q[0] = (1ull << 16) | (d0 << 48);          // data_size=1 (2B), wg_mask=0
  g1.q[1] = (d0 >> 16) | (d1 << 16) | ((unsigned long long)tile_k << 48); // tile_dim0
  g1.q[2] = (unsigned long long)tile_rows | (st0 << 32);                  // tile_dim1
  g1.q[3] = (st0 >> 32);                        // stride1=0
  v4i gz4 = {0, 0, 0, 0};
  v8i gz8 = {0, 0, 0, 0, 0, 0, 0, 0};
  __builtin_amdgcn_tensor_load_to_lds(g0, g1.v, gz4, gz4, gz8, 0);
}

// ---------------------------------------------------------------- WMMA GEMM
// C[M,N] (f32) = A[M,K] (bf16 row-major) * W[N,K]^T (bf16 row-major) [+bias][act]
// act: 0 = none, 1 = gelu, 2 = softplus   (applied after bias add)
// Block: 128 threads = 4 waves; block tile 64(M) x 64(N); K step 32.
// Tiles fetched by the Tensor Data Mover, double-buffered in LDS.
__global__ __launch_bounds__(128) void k_gemm_bf16(
    const __bf16* __restrict__ A, const __bf16* __restrict__ Bw,
    float* __restrict__ C, const float* __restrict__ bias,
    int M, int N, int K, int act)
{
  __shared__ __bf16 sA[2][64 * 32];
  __shared__ __bf16 sB[2][64 * 32];
  const int m0   = blockIdx.y * 64;
  const int n0   = blockIdx.x * 64;
  const int wave = threadIdx.x >> 5;
  const int lane = threadIdx.x & 31;
  const int nK   = (K + 31) >> 5;

  v8f acc0 = v8f_zero(), acc1 = v8f_zero(), acc2 = v8f_zero(), acc3 = v8f_zero();

  if (wave == 0) {   // wave-uniform guard: one wave drives the TDM
    tdm_load_2d((unsigned)(size_t)&sA[0][0], A + (size_t)m0 * K, 64, 32,
                (unsigned)(M - m0), (unsigned)K, (unsigned)K);
    tdm_load_2d((unsigned)(size_t)&sB[0][0], Bw + (size_t)n0 * K, 64, 32,
                (unsigned)(N - n0), (unsigned)K, (unsigned)K);
  }
  int cur = 0;
  for (int kt = 0; kt < nK; ++kt) {
    if (wave == 0) {
      if (kt + 1 < nK) {
        const int k1 = (kt + 1) << 5;
        tdm_load_2d((unsigned)(size_t)&sA[cur ^ 1][0], A + (size_t)m0 * K + k1, 64, 32,
                    (unsigned)(M - m0), (unsigned)(K - k1), (unsigned)K);
        tdm_load_2d((unsigned)(size_t)&sB[cur ^ 1][0], Bw + (size_t)n0 * K + k1, 64, 32,
                    (unsigned)(N - n0), (unsigned)(K - k1), (unsigned)K);
        __builtin_amdgcn_s_wait_tensorcnt(2);   // current buffer's 2 loads done
      } else {
        __builtin_amdgcn_s_wait_tensorcnt(0);
      }
    }
    __syncthreads();
    const __bf16* As = &sA[cur][0];
    const __bf16* Bs = &sB[cur][0];

    // A fragment, 16x32 bf16, ISA 7.12.2 layout.
    v16bf af;
    {
      const int m  = (wave << 4) + (lane & 15);
      const int kb = (lane >> 4) << 3;          // +8 K for upper half-wave
      #pragma unroll
      for (int e = 0; e < 16; ++e) {
        const int v = e >> 1;
        const int k = ((v < 4) ? (v << 1) : (16 + ((v - 4) << 1))) + kb + (e & 1);
        af[e] = As[m * 32 + k];
      }
    }
    // B fragments, 32x16 bf16 each: lanes 0-15 K=0..7,16..23, lanes 16-31 K=8..15,24..31.
    {
      const int koff = (lane >> 4) << 3;
      const int nl   = lane & 15;
      v16bf b0, b1, b2, b3;
      #pragma unroll
      for (int e = 0; e < 16; ++e) {
        const int k = ((e < 8) ? e : (e + 8)) + koff;
        b0[e] = Bs[( 0 + nl) * 32 + k];
        b1[e] = Bs[(16 + nl) * 32 + k];
        b2[e] = Bs[(32 + nl) * 32 + k];
        b3[e] = Bs[(48 + nl) * 32 + k];
      }
      acc0 = __builtin_amdgcn_wmma_f32_16x16x32_bf16(false, af, false, b0, (short)0, acc0, false, false);
      acc1 = __builtin_amdgcn_wmma_f32_16x16x32_bf16(false, af, false, b1, (short)0, acc1, false, false);
      acc2 = __builtin_amdgcn_wmma_f32_16x16x32_bf16(false, af, false, b2, (short)0, acc2, false, false);
      acc3 = __builtin_amdgcn_wmma_f32_16x16x32_bf16(false, af, false, b3, (short)0, acc3, false, false);
    }
    __syncthreads();
    cur ^= 1;
  }

  // Epilogue: C/D layout = VGPR r holds M=r (lanes 0-15) / M=r+8 (lanes 16-31).
  const int half = lane >> 4;
  const int cl   = lane & 15;
  v8f accs[4] = {acc0, acc1, acc2, acc3};
  for (int t = 0; t < 4; ++t) {
    const int col = n0 + (t << 4) + cl;
    if (col >= N) continue;
    const float bv = bias ? bias[col] : 0.f;
    #pragma unroll
    for (int r = 0; r < 8; ++r) {
      const int row = m0 + (wave << 4) + (half << 3) + r;
      if (row < M) {
        float v = accs[t][r] + bv;
        if (act == 1)      v = geluf(v);
        else if (act == 2) v = softplusf(v);
        C[(size_t)row * N + col] = v;
      }
    }
  }
}

// ---------------------------------------------------------------- elementwise
// Assemble (1,20,3,224,224) -> token-major [896*1120, 3]
__global__ void k_assemble(const float* __restrict__ diff, float* __restrict__ out)
{
  long long idx = (long long)blockIdx.x * blockDim.x + threadIdx.x;
  const long long total = 896LL * 1120 * 3;
  if (idx >= total) return;
  int c = (int)(idx % 3);
  long long pos = idx / 3;
  int X = (int)(pos % 1120), Y = (int)(pos / 1120);
  int gh = Y / 224, y = Y % 224, gw = X / 224, x = X % 224;
  int v = gh * 5 + gw;
  out[idx] = diff[(((long long)v * 3 + c) * 224 + y) * 224 + x];
}

// Dense 3x3 conv, pad 1, token-major layout, weight OIHW.
__global__ void k_conv3x3_lc(const float* __restrict__ in, const float* __restrict__ w,
                             const float* __restrict__ bias, float* __restrict__ out,
                             int Cin, int Cout, int Hin, int Win, int stride,
                             int Hout, int Wout)
{
  long long idx = (long long)blockIdx.x * blockDim.x + threadIdx.x;
  long long total = (long long)Hout * Wout * Cout;
  if (idx >= total) return;
  int co = (int)(idx % Cout);
  long long pos = idx / Cout;
  int xo = (int)(pos % Wout), yo = (int)(pos / Wout);
  float s = bias ? bias[co] : 0.f;
  for (int ky = 0; ky < 3; ++ky) {
    int yi = yo * stride + ky - 1;
    if (yi < 0 || yi >= Hin) continue;
    for (int kx = 0; kx < 3; ++kx) {
      int xi = xo * stride + kx - 1;
      if (xi < 0 || xi >= Win) continue;
      const float* ip = in + ((long long)yi * Win + xi) * Cin;
      const float* wp = w + ((long long)co * Cin) * 9 + ky * 3 + kx;
      for (int ci = 0; ci < Cin; ++ci) s += ip[ci] * wp[(long long)ci * 9];
    }
  }
  out[idx] = s;
}

// Channel LayerNorm on token-major [L,C]; in-place safe; optional fused GELU.
__global__ void k_ln_lc(const float* __restrict__ in, const float* __restrict__ g,
                        const float* __restrict__ b, float* __restrict__ out,
                        long long L, int C, int fuse_gelu)
{
  long long t = (long long)blockIdx.x * blockDim.x + threadIdx.x;
  if (t >= L) return;
  const float* x = in + t * C;
  float m = 0.f;
  for (int c = 0; c < C; ++c) m += x[c];
  m /= (float)C;
  float v = 0.f;
  for (int c = 0; c < C; ++c) { float d = x[c] - m; v += d * d; }
  v /= (float)C;
  float r = rsqrtf(v + 1e-5f);
  float* o = out + t * C;
  for (int c = 0; c < C; ++c) {
    float y = (x[c] - m) * r * g[c] + b[c];
    if (fuse_gelu) y = geluf(y);
    o[c] = y;
  }
}

__global__ void k_cvt_bf16(const float* __restrict__ in, __bf16* __restrict__ out,
                           long long n)
{
  long long i = (long long)blockIdx.x * blockDim.x + threadIdx.x;
  if (i < n) out[i] = f2bf(in[i]);
}

// Slice [:, :R] out of [L, Cin] and convert to bf16 [L, R]
__global__ void k_slice_cvt_bf16(const float* __restrict__ in, __bf16* __restrict__ out,
                                 long long L, int R, int Cin)
{
  long long i = (long long)blockIdx.x * blockDim.x + threadIdx.x;
  if (i >= L * R) return;
  long long l = i / R; int r = (int)(i % R);
  out[i] = f2bf(in[l * Cin + r]);
}

// Depthwise 3x3 + SiLU on token-major [L,C]
__global__ void k_dwconv_silu(const float* __restrict__ in, const float* __restrict__ w,
                              float* __restrict__ out, int C, int H, int W)
{
  long long idx = (long long)blockIdx.x * blockDim.x + threadIdx.x;
  long long total = (long long)H * W * C;
  if (idx >= total) return;
  int c = (int)(idx % C);
  long long pos = idx / C;
  int x = (int)(pos % W), y = (int)(pos / W);
  float s = 0.f;
  for (int ky = 0; ky < 3; ++ky) {
    int yi = y + ky - 1;
    if (yi < 0 || yi >= H) continue;
    for (int kx = 0; kx < 3; ++kx) {
      int xi = x + kx - 1;
      if (xi < 0 || xi >= W) continue;
      s += in[((long long)yi * W + xi) * C + c] * w[(long long)c * 9 + ky * 3 + kx];
    }
  }
  out[idx] = siluf(s);
}

// Selective scan, one direction. dir bit0 = transposed order, bit1 = reversed.
// dbl[L, R+2] (cols R / R+1 are B / C), dt[L, C] (softplus'd), x[L, C] (silu'd).
__global__ void k_scan(const float* __restrict__ dbl, const float* __restrict__ dt,
                       const float* __restrict__ x, const float* __restrict__ A_log,
                       const float* __restrict__ Ds, float* __restrict__ yacc,
                       int C, int H, int W, int dir, int accum, int R)
{
  int d = blockIdx.x * blockDim.x + threadIdx.x;
  if (d >= C) return;
  const long long L = (long long)H * W;
  const float Aa = -expf(A_log[d]);
  const float Dd = Ds[d];
  const int Rp2 = R + 2;
  float h = 0.f;
  for (long long t = 0; t < L; ++t) {
    long long tt = (dir & 2) ? (L - 1 - t) : t;
    long long tok;
    if (dir & 1) { long long xc = tt / H, yr = tt % H; tok = yr * W + xc; }
    else tok = tt;
    const float dtv = dt[tok * C + d];
    const float Bv  = dbl[tok * Rp2 + R];
    const float Cv  = dbl[tok * Rp2 + R + 1];
    const float xv  = x[tok * C + d];
    const float a   = expf(dtv * Aa);
    h = a * h + dtv * Bv * xv;
    const float yv = Cv * h + Dd * xv;
    if (accum) yacc[tok * C + d] += yv;
    else       yacc[tok * C + d]  = yv;
  }
}

__global__ void k_add(float* __restrict__ io, const float* __restrict__ a, long long n)
{
  long long i = (long long)blockIdx.x * blockDim.x + threadIdx.x;
  if (i < n) io[i] += a[i];
}

__global__ void k_meanpool(const float* __restrict__ in, float* __restrict__ out,
                           long long L, int C)
{
  int c = blockIdx.x * blockDim.x + threadIdx.x;
  if (c >= C) return;
  float s = 0.f;
  for (long long l = 0; l < L; ++l) s += in[l * C + c];
  out[c] = s / (float)L;
}

// Final head: v[768] @ proj_w^T + proj_b -> gelu -> LayerNorm(512) -> out
__global__ __launch_bounds__(512) void k_head(const float* __restrict__ v,
                                              const float* __restrict__ pw,
                                              const float* __restrict__ pb,
                                              const float* __restrict__ g,
                                              const float* __restrict__ b,
                                              float* __restrict__ out)
{
  __shared__ float sv[512];
  __shared__ float mv[2];
  int i = threadIdx.x;
  float s = pb[i];
  const float* wr = pw + (long long)i * 768;
  for (int k = 0; k < 768; ++k) s += v[k] * wr[k];
  s = geluf(s);
  sv[i] = s;
  __syncthreads();
  if (i == 0) {
    float m = 0.f;
    for (int k = 0; k < 512; ++k) m += sv[k];
    m /= 512.f;
    float var = 0.f;
    for (int k = 0; k < 512; ++k) { float d = sv[k] - m; var += d * d; }
    mv[0] = m; mv[1] = var / 512.f;
  }
  __syncthreads();
  out[i] = (sv[i] - mv[0]) * rsqrtf(mv[1] + 1e-5f) * g[i] + b[i];
}

// ---------------------------------------------------------------- host side
namespace {

struct Bump {
  char* base; size_t off, cap;
  void* get(size_t bytes) {
    void* r = base + off;
    off = (off + bytes + 255) & ~(size_t)255;
    return r;
  }
};

struct BlockP {
  const float *A_log, *Ds, *conv_w, *dt_b, *dt_w, *fc1_b, *fc1_w, *fc2_b, *fc2_w,
              *in_w, *ln1_b, *ln1_g, *ln2_b, *ln2_g, *onorm_b, *onorm_g, *out_w,
              *xproj_w;
};

inline unsigned gb(long long n, int b = 256) { return (unsigned)((n + b - 1) / b); }

inline void gemm(hipStream_t st, const __bf16* A, const __bf16* B, float* C,
                 const float* bias, int M, int N, int K, int act)
{
  dim3 grid((N + 63) / 64, (M + 63) / 64);
  k_gemm_bf16<<<grid, dim3(128), 0, st>>>(A, B, C, bias, M, N, K, act);
}

inline long long llmin(long long a, long long b) { return a < b ? a : b; }

void run_block(hipStream_t st, Bump& ws, float* x, const BlockP& p,
               int C, int H, int W)
{
  const long long L = (long long)H * W;
  const int R = (C + 15) / 16;
  const int Rp2 = R + 2;
  const int Cm = 4 * C;
  const size_t save = ws.off;

  float*  t_ln    = (float*) ws.get((size_t)L * C * 4);
  __bf16* t_bf    = (__bf16*)ws.get((size_t)L * C * 2);
  float*  t_y     = (float*) ws.get((size_t)L * C * 4);
  float*  t_conv  = (float*) ws.get((size_t)L * C * 4);
  __bf16* t_xbf   = (__bf16*)ws.get((size_t)L * C * 2);
  float*  t_dbl   = (float*) ws.get((size_t)L * Rp2 * 4);
  __bf16* t_dtsbf = (__bf16*)ws.get((size_t)L * R * 2);
  float*  t_dt    = (float*) ws.get((size_t)L * C * 4);
  float*  t_yacc  = (float*) ws.get((size_t)L * C * 4);
  __bf16* wbf     = (__bf16*)ws.get((size_t)Cm * C * 2);
  __bf16* wbf2    = (__bf16*)ws.get((size_t)Cm * C * 2);
  const long long CH  = llmin(16384, L);
  float*  t_h     = (float*) ws.get((size_t)CH * Cm * 4);
  __bf16* t_hbf   = (__bf16*)ws.get((size_t)CH * Cm * 2);

  // ---- SS2D branch: ln1 -> in_w -> dwconv+silu -> 4x(xproj,dt,scan) -> onorm -> out_w
  k_ln_lc<<<gb(L), 256, 0, st>>>(x, p.ln1_g, p.ln1_b, t_ln, L, C, 0);
  k_cvt_bf16<<<gb(L * C), 256, 0, st>>>(t_ln, t_bf, L * C);
  k_cvt_bf16<<<gb((long long)C * C), 256, 0, st>>>(p.in_w, wbf, (long long)C * C);
  gemm(st, t_bf, wbf, t_y, nullptr, (int)L, C, C, 0);
  k_dwconv_silu<<<gb(L * C), 256, 0, st>>>(t_y, p.conv_w, t_conv, C, H, W);
  k_cvt_bf16<<<gb(L * C), 256, 0, st>>>(t_conv, t_xbf, L * C);

  for (int k = 0; k < 4; ++k) {
    k_cvt_bf16<<<gb((long long)Rp2 * C), 256, 0, st>>>(p.xproj_w + (size_t)k * Rp2 * C,
                                                       wbf, (long long)Rp2 * C);
    gemm(st, t_xbf, wbf, t_dbl, nullptr, (int)L, Rp2, C, 0);
    k_slice_cvt_bf16<<<gb(L * R), 256, 0, st>>>(t_dbl, t_dtsbf, L, R, Rp2);
    k_cvt_bf16<<<gb((long long)C * R), 256, 0, st>>>(p.dt_w + (size_t)k * C * R,
                                                     wbf, (long long)C * R);
    gemm(st, t_dtsbf, wbf, t_dt, p.dt_b + (size_t)k * C, (int)L, C, R, 2); // softplus
    k_scan<<<(C + 63) / 64, 64, 0, st>>>(t_dbl, t_dt, t_conv,
                                         p.A_log + (size_t)k * C, p.Ds + (size_t)k * C,
                                         t_yacc, C, H, W, k, (k > 0) ? 1 : 0, R);
  }

  k_ln_lc<<<gb(L), 256, 0, st>>>(t_yacc, p.onorm_g, p.onorm_b, t_ln, L, C, 0);
  k_cvt_bf16<<<gb(L * C), 256, 0, st>>>(t_ln, t_bf, L * C);
  k_cvt_bf16<<<gb((long long)C * C), 256, 0, st>>>(p.out_w, wbf, (long long)C * C);
  gemm(st, t_bf, wbf, t_y, nullptr, (int)L, C, C, 0);
  k_add<<<gb(L * C), 256, 0, st>>>(x, t_y, L * C);

  // ---- MLP branch: ln2 -> fc1(+bias,gelu) -> fc2(+bias), token-chunked
  k_ln_lc<<<gb(L), 256, 0, st>>>(x, p.ln2_g, p.ln2_b, t_ln, L, C, 0);
  k_cvt_bf16<<<gb(L * C), 256, 0, st>>>(t_ln, t_bf, L * C);
  k_cvt_bf16<<<gb((long long)Cm * C), 256, 0, st>>>(p.fc1_w, wbf, (long long)Cm * C);
  k_cvt_bf16<<<gb((long long)C * Cm), 256, 0, st>>>(p.fc2_w, wbf2, (long long)C * Cm);
  for (long long off = 0; off < L; off += CH) {
    const int m = (int)llmin(CH, L - off);
    gemm(st, t_bf + off * C, wbf, t_h, p.fc1_b, m, Cm, C, 1);     // gelu
    k_cvt_bf16<<<gb((long long)m * Cm), 256, 0, st>>>(t_h, t_hbf, (long long)m * Cm);
    gemm(st, t_hbf, wbf2, t_y + off * C, p.fc2_b, m, C, Cm, 0);
  }
  k_add<<<gb(L * C), 256, 0, st>>>(x, t_y, L * C);

  ws.off = save;
}

} // namespace

extern "C" void kernel_launch(void* const* d_in, const int* in_sizes, int n_in,
                              void* d_out, int out_size, void* d_ws, size_t ws_size,
                              hipStream_t stream)
{
  (void)in_sizes; (void)n_in; (void)out_size; (void)ws_size;
  // Input flattening assumption: JAX pytree flattening, dict keys sorted:
  //   0: diff
  //   1..12:  downs[0..2]{b, bb, g, w}
  //   13: fln_b  14: fln_g  15: outn_b  16: outn_g
  //   17: pe_b1  18: pe_b2  19: pe_bb1  20: pe_bb2  21: pe_g1  22: pe_g2
  //   23: pe_w1  24: pe_w2  25: proj_b  26: proj_w
  //   27..: 14 blocks x 18 arrays {A_log, Ds, conv_w, dt_b, dt_w, fc1_b, fc1_w,
  //          fc2_b, fc2_w, in_w, ln1_b, ln1_g, ln2_b, ln2_g, onorm_b, onorm_g,
  //          out_w, xproj_w}
  #define P(i) ((const float*)d_in[(i)])
  const float* diff = P(0);
  struct { const float *b, *bb, *g, *w; } downs[3];
  for (int i = 0; i < 3; ++i) { downs[i].b = P(1 + 4*i); downs[i].bb = P(2 + 4*i);
                                downs[i].g = P(3 + 4*i); downs[i].w  = P(4 + 4*i); }
  const float *fln_b = P(13), *fln_g = P(14), *outn_b = P(15), *outn_g = P(16);
  const float *pe_b1 = P(17), *pe_b2 = P(18), *pe_bb1 = P(19), *pe_bb2 = P(20);
  const float *pe_g1 = P(21), *pe_g2 = P(22), *pe_w1 = P(23), *pe_w2 = P(24);
  const float *proj_b = P(25), *proj_w = P(26);

  const int depths[4] = {2, 2, 8, 2};
  BlockP blocks[14];
  {
    int bi = 27, t = 0;
    for (int s = 0; s < 4; ++s)
      for (int dd = 0; dd < depths[s]; ++dd, ++t) {
        BlockP& bp = blocks[t];
        bp.A_log = P(bi+0);  bp.Ds    = P(bi+1);  bp.conv_w = P(bi+2);
        bp.dt_b  = P(bi+3);  bp.dt_w  = P(bi+4);  bp.fc1_b  = P(bi+5);
        bp.fc1_w = P(bi+6);  bp.fc2_b = P(bi+7);  bp.fc2_w  = P(bi+8);
        bp.in_w  = P(bi+9);  bp.ln1_b = P(bi+10); bp.ln1_g  = P(bi+11);
        bp.ln2_b = P(bi+12); bp.ln2_g = P(bi+13); bp.onorm_b = P(bi+14);
        bp.onorm_g = P(bi+15); bp.out_w = P(bi+16); bp.xproj_w = P(bi+17);
        bi += 18;
      }
  }
  #undef P

  Bump ws{(char*)d_ws, 0, ws_size};

  // Persistent activation ping-pong buffers sized for stage 0 (62720 x 96).
  const size_t maxLC = (size_t)224 * 280 * 96;
  float* xA = (float*)ws.get(maxLC * 4);
  float* xB = (float*)ws.get(maxLC * 4);

  // ---- patch embed: assemble -> conv(3->48,s2) -> LN+GELU -> conv(48->96,s2) -> LN
  {
    const size_t save = ws.off;
    float* asm_buf = (float*)ws.get((size_t)896 * 1120 * 3 * 4);
    float* c1      = (float*)ws.get((size_t)448 * 560 * 48 * 4);
    k_assemble<<<gb(896LL * 1120 * 3), 256, 0, stream>>>(diff, asm_buf);
    k_conv3x3_lc<<<gb(448LL * 560 * 48), 256, 0, stream>>>(asm_buf, pe_w1, pe_b1, c1,
                                                           3, 48, 896, 1120, 2, 448, 560);
    k_ln_lc<<<gb(448LL * 560), 256, 0, stream>>>(c1, pe_g1, pe_bb1, c1,
                                                 448LL * 560, 48, 1);
    k_conv3x3_lc<<<gb(224LL * 280 * 96), 256, 0, stream>>>(c1, pe_w2, pe_b2, xA,
                                                           48, 96, 448, 560, 2, 224, 280);
    k_ln_lc<<<gb(224LL * 280), 256, 0, stream>>>(xA, pe_g2, pe_bb2, xA,
                                                 224LL * 280, 96, 0);
    ws.off = save;
  }

  // ---- stages
  float* x = xA; float* xalt = xB;
  int C = 96, H = 224, W = 280, t = 0;
  for (int s = 0; s < 4; ++s) {
    for (int dd = 0; dd < depths[s]; ++dd, ++t)
      run_block(stream, ws, x, blocks[t], C, H, W);
    if (s < 3) {
      const int C2 = C * 2, H2 = (H - 1) / 2 + 1, W2 = (W - 1) / 2 + 1;
      const long long L2 = (long long)H2 * W2;
      k_conv3x3_lc<<<gb(L2 * C2), 256, 0, stream>>>(x, downs[s].w, downs[s].b, xalt,
                                                    C, C2, H, W, 2, H2, W2);
      k_ln_lc<<<gb(L2), 256, 0, stream>>>(xalt, downs[s].g, downs[s].bb, xalt, L2, C2, 0);
      float* tmp = x; x = xalt; xalt = tmp;
      C = C2; H = H2; W = W2;
    }
  }

  // ---- head: outn LN -> mean pool -> proj + gelu -> final LN
  {
    const long long L = (long long)H * W;     // 980 tokens, C = 768
    k_ln_lc<<<gb(L), 256, 0, stream>>>(x, outn_g, outn_b, x, L, C, 0);
    float* v = (float*)ws.get((size_t)C * 4);
    k_meanpool<<<(C + 63) / 64, 64, 0, stream>>>(x, v, L, C);
    k_head<<<1, 512, 0, stream>>>(v, proj_w, proj_b, fln_g, fln_b, (float*)d_out);
  }
}
